// Mlp_13752485282388
// MI455X (gfx1250) — matrix operand order repaired and checked by reading, not compiled, not run on hardware
//
#include <hip/hip_runtime.h>
#include <hip/hip_bf16.h>
#include <math.h>

// ---------------------------------------------------------------------------
// MoE SwiGLU MLP for MI455X (gfx1250, wave32, WMMA bf16 16x16x32)
//   B=2, S=1024 -> T=2048 tokens, H=1024, E=8 experts, F=2816, top-k=2
// router -> per-expert lists -> LDS-staged grouped gate/up GEMM (+SiLU)
//        -> grouped down GEMM (async A-tile copy to LDS) + atomic combine.
// ---------------------------------------------------------------------------

#define HDIM  1024
#define NEXP  8
#define FDIM  2816
#define KSTEP 64
#define BPAD  8
#define LSTR  (KSTEP + BPAD)   // padded LDS k-stride (72 ushorts) vs bank conflicts

typedef __attribute__((ext_vector_type(16))) __bf16          v16bf;
typedef __attribute__((ext_vector_type(16))) unsigned short  v16us;
typedef __attribute__((ext_vector_type(8)))  float           v8f;
typedef __attribute__((ext_vector_type(4)))  float           f32x4;
typedef __attribute__((ext_vector_type(8)))  unsigned short  us8;

#if __has_builtin(__builtin_amdgcn_rcpf)
#define FAST_RCP(x) __builtin_amdgcn_rcpf(x)
#else
#define FAST_RCP(x) (1.0f / (x))
#endif

static __device__ __forceinline__ unsigned short f2bf(float f) {
  __bf16 h = (__bf16)f;                 // native cvt where available (RTNE)
  return __builtin_bit_cast(unsigned short, h);
}
static __device__ __forceinline__ unsigned packbf2(float lo, float hi) {
  return (unsigned)f2bf(lo) | ((unsigned)f2bf(hi) << 16);
}
// Assemble a 16-element bf16 WMMA fragment from two 8-ushort chunks.
static __device__ __forceinline__ v16bf ld_frag16(const unsigned short* p0,
                                                  const unsigned short* p1) {
  us8 a = *(const us8*)p0;
  us8 b = *(const us8*)p1;
  v16us r;
#pragma unroll
  for (int i = 0; i < 8; ++i) { r[i] = a[i]; r[i + 8] = b[i]; }
  return __builtin_bit_cast(v16bf, r);
}

// Async global->LDS copy (4 bytes per lane), tracked by ASYNCcnt.
static __device__ __forceinline__ void async_copy_b32(unsigned lds_byte_addr,
                                                      unsigned long long gaddr) {
  asm volatile("global_load_async_to_lds_b32 %0, %1, off"
               :: "v"(lds_byte_addr), "v"(gaddr) : "memory");
}
static __device__ __forceinline__ void wait_async0() {
#if __has_builtin(__builtin_amdgcn_s_wait_asynccnt)
  __builtin_amdgcn_s_wait_asynccnt(0);
#else
  asm volatile("s_wait_asynccnt 0x0" ::: "memory");
#endif
}

// ---------------------------------------------------------------------------
// Router: one wave32 per token; 8 logits, fp32 softmax, top-2, renorm,
// atomic append (token, weight) into per-expert lists.
// ---------------------------------------------------------------------------
__global__ __launch_bounds__(128)
void router_kernel(const float* __restrict__ x,
                   const float* __restrict__ gate_w,
                   int*   __restrict__ counts,
                   int*   __restrict__ tok_list,
                   float* __restrict__ wt_list,
                   int T) {
  const int lane = threadIdx.x & 31;
  const int wave = threadIdx.x >> 5;
  const int t = blockIdx.x * 4 + wave;
  if (t >= T) return;

  float acc[NEXP];
#pragma unroll
  for (int e = 0; e < NEXP; ++e) acc[e] = 0.f;

  const float* xr = x + (size_t)t * HDIM;
  for (int j = lane; j < HDIM; j += 32) {
    const float xv = xr[j];
    const float* gr = gate_w + (size_t)j * NEXP;
#pragma unroll
    for (int e = 0; e < NEXP; ++e) acc[e] += xv * gr[e];
  }
#pragma unroll
  for (int e = 0; e < NEXP; ++e) {
#pragma unroll
    for (int off = 16; off > 0; off >>= 1)
      acc[e] += __shfl_xor(acc[e], off, 32);
  }

  if (lane == 0) {
    float m = acc[0];
#pragma unroll
    for (int e = 1; e < NEXP; ++e) m = fmaxf(m, acc[e]);
    float p[NEXP], s = 0.f;
#pragma unroll
    for (int e = 0; e < NEXP; ++e) { p[e] = __expf(acc[e] - m); s += p[e]; }
    const float inv = 1.f / s;
#pragma unroll
    for (int e = 0; e < NEXP; ++e) p[e] *= inv;

    int e1 = 0; float p1 = p[0];
#pragma unroll
    for (int e = 1; e < NEXP; ++e) if (p[e] > p1) { p1 = p[e]; e1 = e; }
    int e2 = -1; float p2 = -1.f;
#pragma unroll
    for (int e = 0; e < NEXP; ++e) if (e != e1 && p[e] > p2) { p2 = p[e]; e2 = e; }

    const float norm = 1.f / (p1 + p2);   // NORM_TOPK
    int s1 = atomicAdd(&counts[e1], 1);
    tok_list[e1 * T + s1] = t;
    wt_list [e1 * T + s1] = p1 * norm;
    int s2 = atomicAdd(&counts[e2], 1);
    tok_list[e2 * T + s2] = t;
    wt_list [e2 * T + s2] = p2 * norm;
  }
}

// ---------------------------------------------------------------------------
// Grouped gate/up GEMM + SiLU. Block = 256 thr = 8 waves.
// Tile [16 tokens x 128 F]; 64-K panels staged in LDS as bf16 (A row-major,
// B col-major, padded stride); 4 independent WMMAs per panel.
// ---------------------------------------------------------------------------
__global__ __launch_bounds__(256)
void gateup_kernel(const float* __restrict__ x,
                   const float* __restrict__ w_gate,
                   const float* __restrict__ w_up,
                   const int*   __restrict__ counts,
                   const int*   __restrict__ tok_list,
                   unsigned short* __restrict__ hcur,
                   int T) {
  __shared__ unsigned short lds_a[16 * LSTR];    // x tile   [row][k]
  __shared__ unsigned short lds_g[128 * LSTR];   // gate W   [n][k]
  __shared__ unsigned short lds_u[128 * LSTR];   // up   W   [n][k]

  const int mtiles = T >> 4;
  const int e  = blockIdx.x / mtiles;
  const int mt = blockIdx.x % mtiles;
  const int cnt = counts[e];
  if (mt * 16 >= cnt) return;                    // uniform per block

  const int tid  = threadIdx.x;
  const int lane = tid & 31;
  const int wave = tid >> 5;
  const int n    = lane & 15;
  const int hi   = lane >> 4;
  const int fblk = blockIdx.y * 128;

  // A staging: thread -> (row, 4 consecutive k)
  const int arow = tid >> 4;
  const int ak0  = (tid & 15) * 4;
  const int aslot = min(mt * 16 + arow, cnt - 1);
  const float* xa = x + (size_t)tok_list[e * T + aslot] * HDIM;

  const float* srcG = w_gate + (size_t)e * HDIM * FDIM + fblk;
  const float* srcU = w_up   + (size_t)e * HDIM * FDIM + fblk;

  // B staging: column fixed per thread, k walks by 4 (256 thr = 2 k-pair rows)
  const int bn  = tid & 127;
  const int bk0 = (tid >> 7) * 2;                // 0 or 2
  const float* gcol = srcG + bn + (size_t)bk0 * FDIM;
  const float* ucol = srcU + bn + (size_t)bk0 * FDIM;
  unsigned short* lgb = &lds_g[bn * LSTR + bk0];
  unsigned short* lub = &lds_u[bn * LSTR + bk0];

  v8f accg0 = {}, accg1 = {}, accu0 = {}, accu1 = {};

  for (int kb = 0; kb < HDIM; kb += KSTEP) {
    // ---- stage A: 16x64 fp32 -> bf16 (coalesced 4-float read per thread) --
    {
      f32x4 v = *(const f32x4*)(xa + kb + ak0);
      unsigned* dst = (unsigned*)&lds_a[arow * LSTR + ak0];
      dst[0] = packbf2(v[0], v[1]);
      dst[1] = packbf2(v[2], v[3]);
    }
    // ---- stage B: 64x128 fp32 -> bf16 col-major, running row pointers -----
    {
      const float* pg = gcol + (size_t)kb * FDIM;
      const float* pu = ucol + (size_t)kb * FDIM;
#pragma unroll
      for (int rep = 0; rep < 16; ++rep) {
        *(unsigned*)&lgb[rep * 4] = packbf2(pg[0], pg[FDIM]);
        *(unsigned*)&lub[rep * 4] = packbf2(pu[0], pu[FDIM]);
        pg += 4 * (size_t)FDIM;
        pu += 4 * (size_t)FDIM;
      }
    }
    if (kb + KSTEP < HDIM) {
      __builtin_prefetch(srcG + (size_t)(kb + KSTEP) * FDIM + bn, 0, 1);
      __builtin_prefetch(srcU + (size_t)(kb + KSTEP) * FDIM + bn, 0, 1);
    }
    __syncthreads();

    const int col = wave * 16 + n;
    {
      const unsigned short* ap0 = &lds_a[n * LSTR + hi * 16];
      const unsigned short* gp0 = &lds_g[col * LSTR + hi * 16];
      const unsigned short* up0 = &lds_u[col * LSTR + hi * 16];
      v16bf a0 = ld_frag16(ap0,      ap0 + 8);
      v16bf a1 = ld_frag16(ap0 + 32, ap0 + 40);
      v16bf g0 = ld_frag16(gp0,      gp0 + 8);
      v16bf g1 = ld_frag16(gp0 + 32, gp0 + 40);
      v16bf u0 = ld_frag16(up0,      up0 + 8);
      v16bf u1 = ld_frag16(up0 + 32, up0 + 40);
      accg0 = __builtin_amdgcn_wmma_f32_16x16x32_bf16(false, a0, false, g0, (short)0, accg0, false, false);
      accu0 = __builtin_amdgcn_wmma_f32_16x16x32_bf16(false, a0, false, u0, (short)0, accu0, false, false);
      accg1 = __builtin_amdgcn_wmma_f32_16x16x32_bf16(false, a1, false, g1, (short)0, accg1, false, false);
      accu1 = __builtin_amdgcn_wmma_f32_16x16x32_bf16(false, a1, false, u1, (short)0, accu1, false, false);
    }
    __syncthreads();
  }

  const v8f accg = accg0 + accg1;
  const v8f accu = accu0 + accu1;

  // D layout: VGPR r -> row M = r + 8*hi, col N = lane%16. SiLU(g)*u -> bf16.
#pragma unroll
  for (int r = 0; r < 8; ++r) {
    const int m = r + hi * 8;
    const int oslot = mt * 16 + m;
    if (oslot < cnt) {
      const float g = accg[r], u = accu[r];
      const float h = g * FAST_RCP(1.f + __expf(-g)) * u;
      hcur[((size_t)e * T + oslot) * FDIM + fblk + wave * 16 + n] = f2bf(h);
    }
  }
}

// ---------------------------------------------------------------------------
// Grouped down GEMM + weighted combine. Tile [16 pairs x 128 H].
// A tile (hcur, already bf16) copied global->LDS with ASYNC loads overlapped
// with the fp32->bf16 B staging; both consumed via ds_load_b128 fragments.
// ---------------------------------------------------------------------------
__global__ __launch_bounds__(256)
void down_kernel(const unsigned short* __restrict__ hcur,
                 const float* __restrict__ w_down,
                 const int*   __restrict__ counts,
                 const int*   __restrict__ tok_list,
                 const float* __restrict__ wt_list,
                 float* __restrict__ out,
                 int T) {
  __shared__ unsigned short lds_a[16 * LSTR];    // hcur tile [row][k]
  __shared__ unsigned short lds_b[128 * LSTR];   // down W    [n][k]

  const int mtiles = T >> 4;
  const int e  = blockIdx.x / mtiles;
  const int mt = blockIdx.x % mtiles;
  const int cnt = counts[e];
  if (mt * 16 >= cnt) return;

  const int tid  = threadIdx.x;
  const int lane = tid & 31;
  const int wave = tid >> 5;
  const int n    = lane & 15;
  const int hi   = lane >> 4;
  const int hblk = blockIdx.y * 128;

  // A staging: thread -> rows (tid>>5) and (tid>>5)+8, one uint (2 bf16) each
  const int arow = tid >> 5;
  const int au   = tid & 31;
  const int as0 = min(mt * 16 + arow,     cnt - 1);
  const int as1 = min(mt * 16 + arow + 8, cnt - 1);
  const unsigned* pa0 = (const unsigned*)(hcur + ((size_t)e * T + as0) * FDIM) + au;
  const unsigned* pa1 = (const unsigned*)(hcur + ((size_t)e * T + as1) * FDIM) + au;
  const unsigned a_lds0 = (unsigned)(size_t)&lds_a[arow * LSTR] + au * 4u;
  const unsigned a_lds1 = (unsigned)(size_t)&lds_a[(arow + 8) * LSTR] + au * 4u;

  const float* srcD = w_down + (size_t)e * FDIM * HDIM + hblk;
  const int bn  = tid & 127;
  const int bk0 = (tid >> 7) * 2;
  const float* dcol = srcD + bn + (size_t)bk0 * HDIM;
  unsigned short* lbb = &lds_b[bn * LSTR + bk0];

  v8f acc0 = {}, acc1 = {};
  for (int kb = 0; kb < FDIM; kb += KSTEP) {
    // ---- stage A: async byte-copy global->LDS (ASYNCcnt) ------------------
    async_copy_b32(a_lds0, (unsigned long long)(size_t)pa0);
    async_copy_b32(a_lds1, (unsigned long long)(size_t)pa1);
    pa0 += KSTEP / 2;
    pa1 += KSTEP / 2;
    // ---- stage B: 64x128 fp32 -> bf16 col-major (overlaps async) ----------
    {
      const float* pd = dcol + (size_t)kb * HDIM;
#pragma unroll
      for (int rep = 0; rep < 16; ++rep) {
        *(unsigned*)&lbb[rep * 4] = packbf2(pd[0], pd[HDIM]);
        pd += 4 * (size_t)HDIM;
      }
    }
    if (kb + KSTEP < FDIM)
      __builtin_prefetch(srcD + (size_t)(kb + KSTEP) * HDIM + bn, 0, 1);
    wait_async0();
    __syncthreads();

    const int col = wave * 16 + n;
    {
      const unsigned short* ap0 = &lds_a[n * LSTR + hi * 16];
      const unsigned short* bp0 = &lds_b[col * LSTR + hi * 16];
      v16bf a0 = ld_frag16(ap0,      ap0 + 8);
      v16bf a1 = ld_frag16(ap0 + 32, ap0 + 40);
      v16bf b0 = ld_frag16(bp0,      bp0 + 8);
      v16bf b1 = ld_frag16(bp0 + 32, bp0 + 40);
      acc0 = __builtin_amdgcn_wmma_f32_16x16x32_bf16(false, a0, false, b0, (short)0, acc0, false, false);
      acc1 = __builtin_amdgcn_wmma_f32_16x16x32_bf16(false, a1, false, b1, (short)0, acc1, false, false);
    }
    __syncthreads();
  }
  const v8f acc = acc0 + acc1;

  // Two experts contribute per token -> f32 global atomic add.
#pragma unroll
  for (int r = 0; r < 8; ++r) {
    const int m = r + hi * 8;
    const int oslot = mt * 16 + m;
    if (oslot < cnt) {
      const int   tok = tok_list[e * T + oslot];
      const float w   = wt_list [e * T + oslot];
      __hip_atomic_fetch_add(&out[(size_t)tok * HDIM + hblk + wave * 16 + n],
                             w * acc[r],
                             __ATOMIC_RELAXED, __HIP_MEMORY_SCOPE_AGENT);
    }
  }
}

// ---------------------------------------------------------------------------
// Launch. Workspace layout (worst case ~92.5 MB):
//   [0, 32)            : counts[E]
//   [256, +E*T*4)      : tok_list
//   [.., +E*T*4)       : wt_list
//   [.., +E*T*F*2)     : hcur (bf16)
// ---------------------------------------------------------------------------
extern "C" void kernel_launch(void* const* d_in, const int* in_sizes, int n_in,
                              void* d_out, int out_size, void* d_ws, size_t ws_size,
                              hipStream_t stream) {
  const float* x      = (const float*)d_in[0];
  const float* gate_w = (const float*)d_in[1];
  const float* w_gate = (const float*)d_in[2];
  const float* w_up   = (const float*)d_in[3];
  const float* w_down = (const float*)d_in[4];
  float* out = (float*)d_out;
  const int T = in_sizes[0] / HDIM;

  char* ws = (char*)d_ws;
  int*   counts   = (int*)ws;
  int*   tok_list = (int*)(ws + 256);
  float* wt_list  = (float*)(ws + 256 + (size_t)NEXP * T * 4);
  unsigned short* hcur = (unsigned short*)(ws + 256 + 2 * (size_t)NEXP * T * 4);

  hipMemsetAsync(counts, 0, NEXP * sizeof(int), stream);
  hipMemsetAsync(out, 0, (size_t)out_size * sizeof(float), stream);

  router_kernel<<<dim3((T + 3) / 4), dim3(128), 0, stream>>>(
      x, gate_w, counts, tok_list, wt_list, T);

  dim3 gridA(NEXP * (T / 16), FDIM / 128);   // blocks past count[e] exit early
  gateup_kernel<<<gridA, dim3(256), 0, stream>>>(
      x, w_gate, w_up, counts, tok_list, hcur, T);

  dim3 gridB(NEXP * (T / 16), HDIM / 128);
  down_kernel<<<gridB, dim3(256), 0, stream>>>(
      hcur, w_down, counts, tok_list, wt_list, out, T);
}
